// VariationalGAE_49589692399793
// MI455X (gfx1250) — matrix operand order repaired and checked by reading, not compiled, not run on hardware
//
#include <hip/hip_runtime.h>
#include <stdint.h>

// ---------------- constants from the reference ----------------
#define NN   10000   // nodes
#define INC  256     // in_channels
#define HID  128     // 2*out_channels
#define OUTC 64      // out_channels
#define NE   640000  // edges
#define KSPLIT 4
#define KPER   2500  // NN / KSPLIT
#define LPAD 40      // LDS row stride in shorts (80B, 16B-aligned rows)

typedef __bf16 v16bf __attribute__((ext_vector_type(16)));
typedef float  v8f   __attribute__((ext_vector_type(8)));

union Frag { unsigned u[8]; uint4 q[2]; v16bf v; };

__device__ __forceinline__ unsigned short f2bf(float f) {
  unsigned u = __builtin_bit_cast(unsigned, f);
  u += 0x7FFFu + ((u >> 16) & 1u);          // round-to-nearest-even
  return (unsigned short)(u >> 16);
}
__device__ __forceinline__ unsigned pack2bf(float a, float b) {
  return (unsigned)f2bf(a) | ((unsigned)f2bf(b) << 16);
}
__device__ __forceinline__ float bf2f(unsigned short h) {
  unsigned u = ((unsigned)h) << 16;
  return __builtin_bit_cast(float, u);
}
__device__ __forceinline__ v8f wmma_bf16(const Frag& a, const Frag& b, v8f c) {
  return __builtin_amdgcn_wmma_f32_16x16x32_bf16(false, a.v, false, b.v,
                                                 (short)0, c, false, false);
}
__device__ __forceinline__ void atomic_add_f32(float* p, float v) {
  __hip_atomic_fetch_add(p, v, __ATOMIC_RELAXED, __HIP_MEMORY_SCOPE_AGENT);
}

// load a WMMA A-fragment (16x32 bf16) from LDS row `ar`: two contiguous 16B runs
__device__ __forceinline__ void load_afrag(Frag& a, const unsigned short (*As)[LPAD],
                                           int ar, int half) {
  a.q[0] = *(const uint4*)&As[ar][8 * half];
  a.q[1] = *(const uint4*)&As[ar][16 + 8 * half];
}
// load a WMMA B-fragment (32x16 bf16) from transposed LDS col-row `c`
__device__ __forceinline__ void load_bfrag(Frag& b, const unsigned short (*Bs)[LPAD],
                                           int c, int half) {
  b.q[0] = *(const uint4*)&Bs[c][8 * half];
  b.q[1] = *(const uint4*)&Bs[c][16 + 8 * half];
}

// ---------------- utility kernels ----------------
__global__ void zero_f32(float* __restrict__ p, int n) {
  int i = blockIdx.x * blockDim.x + threadIdx.x;
  int stride = gridDim.x * blockDim.x;
  for (; i < n; i += stride) p[i] = 0.f;
}

__global__ void cvt_f32_bf16(const float* __restrict__ s,
                             unsigned short* __restrict__ d, int n) {
  int i = blockIdx.x * blockDim.x + threadIdx.x;
  int stride = gridDim.x * blockDim.x;
  for (; i < n; i += stride) d[i] = f2bf(s[i]);
}

__global__ void deg_kernel(const int* __restrict__ src, float* __restrict__ deg, int e) {
  int i = blockIdx.x * blockDim.x + threadIdx.x;
  if (i < e) atomic_add_f32(&deg[src[i]], 1.0f);
}

__global__ void dinv_kernel(const float* __restrict__ deg, float* __restrict__ dinv, int n) {
  int i = blockIdx.x * blockDim.x + threadIdx.x;
  if (i < n) {
    float dg = deg[i];
    dinv[i] = dg > 0.f ? rsqrtf(fmaxf(dg, 1.f)) : 0.f;
  }
}

__global__ void bias_relu_bf16(const float* __restrict__ hf, const float* __restrict__ b,
                               unsigned short* __restrict__ hb, int n, int cmask) {
  int i = blockIdx.x * blockDim.x + threadIdx.x;
  int stride = gridDim.x * blockDim.x;
  for (; i < n; i += stride)
    hb[i] = f2bf(fmaxf(hf[i] + b[i & cmask], 0.f));
}

// ---------------- big GEMM: h1f += x[128x2500] @ Wt_bf16[2500x128] per block (split-K) ----
__global__ __launch_bounds__(256) void gemm_x_wt(
    const float* __restrict__ x, const unsigned short* __restrict__ Wtb,
    float* __restrict__ h1f) {
  __shared__ __attribute__((aligned(16))) unsigned short As[128][LPAD];  // [row][k]
  __shared__ __attribute__((aligned(16))) unsigned short Bs[128][LPAD];  // [col][k] transposed

  const int tid  = threadIdx.x;
  const int lane = tid & 31;
  const int wave = tid >> 5;
  const int m    = lane & 15;
  const int half = lane >> 4;
  const int rowBase = blockIdx.x * 128;
  const int c0      = blockIdx.y * 128;
  const int k0      = blockIdx.z * KPER;
  const int kend    = min(NN, k0 + KPER);

  v8f acc[8];
  #pragma unroll
  for (int t = 0; t < 8; ++t)
    #pragma unroll
    for (int i = 0; i < 8; ++i) acc[t][i] = 0.f;

  for (int ks = k0; ks < kend; ks += 32) {
    const bool fullK = (ks + 32 <= kend);
    if (fullK) {
      // ---- branchless fast path ----
      #pragma unroll
      for (int i = 0; i < 4; ++i) {
        int idx = tid + i * 256;           // 0..1023 float4 slots
        int r = idx >> 3;
        int q = idx & 7;
        int gr = min(rowBase + r, NN - 1); // OOB rows only feed guarded outputs
        const float* p = x + (size_t)gr * NN + ks + q * 4;
        float4 v = *(const float4*)p;
        __builtin_prefetch(p + 32, 0, 0);  // global_prefetch_b8: next k tile of x
        *(uint2*)&As[r][q * 4] = make_uint2(pack2bf(v.x, v.y), pack2bf(v.z, v.w));
      }
      #pragma unroll
      for (int i = 0; i < 8; ++i) {
        int idx = tid + i * 256;           // 0..2047 uint slots
        int kk = idx >> 6;
        int cu = idx & 63;
        unsigned w = *(const unsigned*)(Wtb + (size_t)(ks + kk) * INC + c0 + cu * 2);
        Bs[cu * 2 + 0][kk] = (unsigned short)(w & 0xffffu);
        Bs[cu * 2 + 1][kk] = (unsigned short)(w >> 16);
      }
    } else {
      // ---- k-tail (last split-K step only) ----
      #pragma unroll
      for (int i = 0; i < 4; ++i) {
        int idx = tid + i * 256;
        int r = idx >> 3;
        int q = idx & 7;
        int gr = min(rowBase + r, NN - 1);
        int gk = ks + q * 4;
        float4 v = make_float4(0.f, 0.f, 0.f, 0.f);
        if (gk + 4 <= kend)
          v = *(const float4*)(x + (size_t)gr * NN + gk);
        *(uint2*)&As[r][q * 4] = make_uint2(pack2bf(v.x, v.y), pack2bf(v.z, v.w));
      }
      #pragma unroll
      for (int i = 0; i < 8; ++i) {
        int idx = tid + i * 256;
        int kk = idx >> 6;
        int cu = idx & 63;
        int gk = ks + kk;
        unsigned w = 0;
        if (gk < kend)
          w = *(const unsigned*)(Wtb + (size_t)gk * INC + c0 + cu * 2);
        Bs[cu * 2 + 0][kk] = (unsigned short)(w & 0xffffu);
        Bs[cu * 2 + 1][kk] = (unsigned short)(w >> 16);
      }
    }
    __syncthreads();

    Frag a;
    load_afrag(a, As, wave * 16 + m, half);
    #pragma unroll
    for (int t = 0; t < 8; ++t) {
      Frag b;
      load_bfrag(b, Bs, t * 16 + m, half);
      acc[t] = wmma_bf16(a, b, acc[t]);
    }
    __syncthreads();
  }

  // split-K accumulate into f32 buffer
  #pragma unroll
  for (int t = 0; t < 8; ++t) {
    int gc = c0 + t * 16 + m;
    #pragma unroll
    for (int v = 0; v < 8; ++v) {
      int gr = rowBase + wave * 16 + half * 8 + v;
      if (gr < NN) atomic_add_f32(&h1f[(size_t)gr * INC + gc], acc[t][v]);
    }
  }
}

// ---------------- edge scatter: tx[dst] += (-dinv[s]*dinv[d]) * h[src] ----------------
template <int C>
__global__ __launch_bounds__(256) void scatter_add(
    const int* __restrict__ src, const int* __restrict__ dst,
    const float* __restrict__ dinv, const unsigned short* __restrict__ h,
    float* __restrict__ tx, int e_count) {
  int e = blockIdx.x * (blockDim.x >> 5) + (threadIdx.x >> 5);
  if (e >= e_count) return;
  int lane = threadIdx.x & 31;
  int s = src[e];
  int d = dst[e];
  float w = -dinv[s] * dinv[d];
  #pragma unroll
  for (int i = 0; i < C / 32; ++i) {
    int c = lane + i * 32;
    float v = w * bf2f(h[(size_t)s * C + c]);
    atomic_add_f32(&tx[(size_t)d * C + c], v);
  }
}

// ---------------- fused Cheb GEMM: out = act(Ha@W0 + Hb@W1 + bias) ----------------
template <int KD, int CD, bool RELU>
__global__ __launch_bounds__(256) void cheb_gemm(
    const unsigned short* __restrict__ Ha,  // [NN][KD] bf16
    const float* __restrict__ Hb,           // [NN][KD] f32
    const unsigned short* __restrict__ W0b, // [KD][CD] bf16
    const unsigned short* __restrict__ W1b, // [KD][CD] bf16
    const float* __restrict__ bias,
    unsigned short* __restrict__ outb,      // used if RELU
    float* __restrict__ outf) {             // used if !RELU
  __shared__ __attribute__((aligned(16))) unsigned short As[128][LPAD];
  __shared__ __attribute__((aligned(16))) unsigned short Bs[CD][LPAD];

  const int tid  = threadIdx.x;
  const int lane = tid & 31;
  const int wave = tid >> 5;
  const int m    = lane & 15;
  const int half = lane >> 4;
  const int rowBase = blockIdx.x * 128;
  constexpr int CT = CD / 16;

  v8f acc[CT];
  #pragma unroll
  for (int t = 0; t < CT; ++t)
    #pragma unroll
    for (int i = 0; i < 8; ++i) acc[t][i] = 0.f;

  #pragma unroll
  for (int pass = 0; pass < 2; ++pass) {
    const unsigned short* W = (pass == 0) ? W0b : W1b;
    for (int ks = 0; ks < KD; ks += 32) {
      if (pass == 0) {  // bf16 source (branchless, rows clamped)
        #pragma unroll
        for (int i = 0; i < 8; ++i) {
          int idx = tid + i * 256;       // 2048 uints = 128x32 bf16
          int r = idx >> 4;
          int ku = idx & 15;
          int gr = min(rowBase + r, NN - 1);
          unsigned w = *(const unsigned*)(Ha + (size_t)gr * KD + ks + ku * 2);
          *(unsigned*)&As[r][ku * 2] = w;
        }
      } else {          // f32 source, convert on load (branchless, rows clamped)
        #pragma unroll
        for (int i = 0; i < 4; ++i) {
          int idx = tid + i * 256;       // 1024 float4 slots
          int r = idx >> 3;
          int q = idx & 7;
          int gr = min(rowBase + r, NN - 1);
          float4 v = *(const float4*)(Hb + (size_t)gr * KD + ks + q * 4);
          *(uint2*)&As[r][q * 4] = make_uint2(pack2bf(v.x, v.y), pack2bf(v.z, v.w));
        }
      }
      // weight tile 32 x CD, transposed into Bs[col][k]
      #pragma unroll
      for (int i = 0; i < CD / 16; ++i) {
        int idx = tid + i * 256;         // 32*CD/2 uints
        int kk = idx / (CD / 2);
        int cu = idx % (CD / 2);
        unsigned w = *(const unsigned*)(W + (size_t)(ks + kk) * CD + cu * 2);
        Bs[cu * 2 + 0][kk] = (unsigned short)(w & 0xffffu);
        Bs[cu * 2 + 1][kk] = (unsigned short)(w >> 16);
      }
      __syncthreads();

      Frag a;
      load_afrag(a, As, wave * 16 + m, half);
      #pragma unroll
      for (int t = 0; t < CT; ++t) {
        Frag b;
        load_bfrag(b, Bs, t * 16 + m, half);
        acc[t] = wmma_bf16(a, b, acc[t]);
      }
      __syncthreads();
    }
  }

  #pragma unroll
  for (int t = 0; t < CT; ++t) {
    int gc = t * 16 + m;
    float bv = bias[gc];
    #pragma unroll
    for (int v = 0; v < 8; ++v) {
      int gr = rowBase + wave * 16 + half * 8 + v;
      if (gr < NN) {
        float val = acc[t][v] + bv;
        if (RELU) outb[(size_t)gr * CD + gc] = f2bf(fmaxf(val, 0.f));
        else      outf[(size_t)gr * CD + gc] = val;
      }
    }
  }
}

// ---------------- host launcher ----------------
extern "C" void kernel_launch(void* const* d_in, const int* in_sizes, int n_in,
                              void* d_out, int out_size, void* d_ws, size_t ws_size,
                              hipStream_t stream) {
  (void)in_sizes; (void)n_in; (void)out_size; (void)ws_size;

  const float* x   = (const float*)d_in[0];
  const int*   ei  = (const int*)d_in[1];
  const float* Wt  = (const float*)d_in[2];
  const float* bt  = (const float*)d_in[3];
  const float* W01 = (const float*)d_in[4];
  const float* W11 = (const float*)d_in[5];
  const float* b1  = (const float*)d_in[6];
  const float* W0m = (const float*)d_in[7];
  const float* W1m = (const float*)d_in[8];
  const float* bm  = (const float*)d_in[9];
  const float* W0l = (const float*)d_in[10];
  const float* W1l = (const float*)d_in[11];
  const float* bl  = (const float*)d_in[12];

  const int* src = ei;
  const int* dst = ei + NE;
  float* mu = (float*)d_out;
  float* ls = (float*)d_out + (size_t)NN * OUTC;

  // workspace layout (zeroed region first, contiguous)
  char* w = (char*)d_ws;
  float* h1f  = (float*)w; w += (size_t)NN * INC * 4;   // 10.24 MB
  float* tx1  = (float*)w; w += (size_t)NN * INC * 4;   // 10.24 MB
  float* tx2  = (float*)w; w += (size_t)NN * HID * 4;   //  5.12 MB
  float* deg  = (float*)w; w += (size_t)NN * 4;
  float* dinv = (float*)w; w += (size_t)NN * 4;
  unsigned short* h1b  = (unsigned short*)w; w += (size_t)NN * INC * 2;
  unsigned short* h2b  = (unsigned short*)w; w += (size_t)NN * HID * 2;
  unsigned short* Wtb  = (unsigned short*)w; w += (size_t)NN * INC * 2;
  unsigned short* W01b = (unsigned short*)w; w += (size_t)INC * HID * 2;
  unsigned short* W11b = (unsigned short*)w; w += (size_t)INC * HID * 2;
  unsigned short* W0mb = (unsigned short*)w; w += (size_t)HID * OUTC * 2;
  unsigned short* W1mb = (unsigned short*)w; w += (size_t)HID * OUTC * 2;
  unsigned short* W0lb = (unsigned short*)w; w += (size_t)HID * OUTC * 2;
  unsigned short* W1lb = (unsigned short*)w; w += (size_t)HID * OUTC * 2;

  const int zeroN = NN * INC + NN * INC + NN * HID + NN;  // h1f+tx1+tx2+deg
  zero_f32<<<4096, 256, 0, stream>>>(h1f, zeroN);

  // weight conversions to bf16
  cvt_f32_bf16<<<4096, 256, 0, stream>>>(Wt,  Wtb,  NN * INC);
  cvt_f32_bf16<<<64,   256, 0, stream>>>(W01, W01b, INC * HID);
  cvt_f32_bf16<<<64,   256, 0, stream>>>(W11, W11b, INC * HID);
  cvt_f32_bf16<<<32,   256, 0, stream>>>(W0m, W0mb, HID * OUTC);
  cvt_f32_bf16<<<32,   256, 0, stream>>>(W1m, W1mb, HID * OUTC);
  cvt_f32_bf16<<<32,   256, 0, stream>>>(W0l, W0lb, HID * OUTC);
  cvt_f32_bf16<<<32,   256, 0, stream>>>(W1l, W1lb, HID * OUTC);

  // edge norm
  deg_kernel<<<(NE + 255) / 256, 256, 0, stream>>>(src, deg, NE);
  dinv_kernel<<<(NN + 255) / 256, 256, 0, stream>>>(deg, dinv, NN);

  // h1 = relu(x @ Wt + bt)  (bf16 WMMA, split-K)
  dim3 g1((NN + 127) / 128, INC / 128, KSPLIT);
  gemm_x_wt<<<g1, 256, 0, stream>>>(x, Wtb, h1f);
  bias_relu_bf16<<<4096, 256, 0, stream>>>(h1f, bt, h1b, NN * INC, INC - 1);

  // tx1 = segment_sum(w * h1[src], dst)
  scatter_add<INC><<<(NE + 7) / 8, 256, 0, stream>>>(src, dst, dinv, h1b, tx1, NE);

  // h2 = relu(h1 @ W0_1 + tx1 @ W1_1 + b1)
  cheb_gemm<INC, HID, true><<<(NN + 127) / 128, 256, 0, stream>>>(
      h1b, tx1, W01b, W11b, b1, h2b, nullptr);

  // tx2 = segment_sum(w * h2[src], dst)
  scatter_add<HID><<<(NE + 7) / 8, 256, 0, stream>>>(src, dst, dinv, h2b, tx2, NE);

  // mu / logstd
  cheb_gemm<HID, OUTC, false><<<(NN + 127) / 128, 256, 0, stream>>>(
      h2b, tx2, W0mb, W1mb, bm, nullptr, mu);
  cheb_gemm<HID, OUTC, false><<<(NN + 127) / 128, 256, 0, stream>>>(
      h2b, tx2, W0lb, W1lb, bl, nullptr, ls);
}
